// CRFOutputLayer_82961588289821
// MI455X (gfx1250) — matrix operand order repaired and checked by reading, not compiled, not run on hardware
//
#include <hip/hip_runtime.h>
#include <hip/hip_bf16.h>
#include <math.h>

typedef __attribute__((ext_vector_type(16))) _Float16 v16h;
typedef __attribute__((ext_vector_type(8)))  _Float16 v8h;
typedef __attribute__((ext_vector_type(8)))  float    v8f;

static constexpr int B = 128;
static constexpr int S = 2048;
static constexpr int T = 128;

#define LOG2E 1.44269504088896340736f
#define LN2   0.69314718055994530942f

// raw v_exp_f32 / v_log_f32 (base-2), no denormal fixup sequences
__device__ __forceinline__ float fast_exp(float x) {           // e^x, x <= ~0
    return __builtin_amdgcn_exp2f(x * LOG2E);
}
__device__ __forceinline__ float fast_log(float x) {           // ln(x), x >= ~1e-30
    return __builtin_amdgcn_logf(x) * LN2;
}

// ---------------------------------------------------------------------------
// Forward (logsumexp) scan via WMMA.
// One block = 16 batch rows, 8 waves; wave w owns tag columns [16w, 16w+16).
// nxt[j] = log( sum_i exp(alpha_i - m) * exp(trans[i][j]) ) + m + e_t[j]
// exp(trans) B-fragments live in VGPRs for the entire 2048-step scan.
// sched_barrier keeps the 8 ds_load_b128 A-fragment loads clustered ahead of
// the WMMA chain so the dscnt waits are staged instead of fully drained.
// ---------------------------------------------------------------------------
__global__ __launch_bounds__(256) void crf_forward(
    const float* __restrict__ logit, const int* __restrict__ seq_lens,
    const float* __restrict__ trans, const float* __restrict__ start_t,
    const float* __restrict__ end_t, float* __restrict__ logZ)
{
    __shared__ __align__(16) _Float16 E16s[T * T];   // exp(trans) f16, 32 KB
    __shared__ __align__(16) _Float16 p16s[16 * T];  // exp(alpha-m) f16, 4 KB
    __shared__ float alpha_s[16 * T];                // 8 KB
    __shared__ float mrow_s[16];
    __shared__ int   len_s[16];

    const int tid = threadIdx.x;
    const int b0  = blockIdx.x * 16;

    for (int q = tid; q < T * T; q += 256) E16s[q] = (_Float16)__expf(trans[q]);
    for (int q = tid; q < 16 * T; q += 256) {
        int m = q >> 7, j = q & 127;
        alpha_s[q] = start_t[j] + logit[((size_t)(b0 + m) * S) * T + j];
    }
    if (tid < 16) len_s[tid] = seq_lens[b0 + tid];
    __syncthreads();

    const int L = tid & 31;       // lane in wave32
    const int w = tid >> 5;       // wave id = column tile
    const int lh = L >> 4;        // lane half
    const int ll = L & 15;

    // B fragments: bf[kc][e] = E[k][n], k = 32*kc + 8*lh + e + 8*(e>>3), n = 16*w + ll
    v16h bf[4];
#pragma unroll
    for (int kc = 0; kc < 4; ++kc)
#pragma unroll
        for (int e = 0; e < 16; ++e)
            bf[kc][e] = E16s[(32 * kc + 8 * lh + e + 8 * (e >> 3)) * T + 16 * w + ll];

    const int m_a = tid >> 4;     // reduction mapping: 16 threads per batch row
    const int jl  = tid & 15;

    for (int t = 1; t < S; ++t) {
        // row max over alpha, then p = exp(alpha - m) into LDS (f16)
        float mx = -INFINITY;
#pragma unroll
        for (int k2 = 0; k2 < 8; ++k2)
            mx = fmaxf(mx, alpha_s[m_a * T + jl + 16 * k2]);
#pragma unroll
        for (int off = 8; off; off >>= 1)
            mx = fmaxf(mx, __shfl_xor(mx, off, 32));
        if (jl == 0) mrow_s[m_a] = mx;
#pragma unroll
        for (int k2 = 0; k2 < 8; ++k2) {
            int j = jl + 16 * k2;
            p16s[m_a * T + j] = (_Float16)fast_exp(alpha_s[m_a * T + j] - mx);
        }
        __syncthreads();

        // emission loads for this wave's tile (issued early, overlap with WMMA)
        const int jj = 16 * w + ll;
        float ev[8];
#pragma unroll
        for (int r = 0; r < 8; ++r) {
            int m = r + 8 * lh;
            ev[r] = logit[((size_t)(b0 + m) * S + t) * T + jj];
        }

        // Preload all 4 A fragments (8 ds_load_b128, kept ahead of the WMMAs)
        v16h afr[4];
#pragma unroll
        for (int kc = 0; kc < 4; ++kc) {
            const _Float16* rowp = &p16s[ll * T + 32 * kc + 8 * lh];
            v8h lo = *(const v8h*)(rowp);        // K = k0..k0+7
            v8h hi = *(const v8h*)(rowp + 16);   // K = k0+16..k0+23
#pragma unroll
            for (int e = 0; e < 8; ++e) { afr[kc][e] = lo[e]; afr[kc][e + 8] = hi[e]; }
        }
        __builtin_amdgcn_sched_barrier(0);       // loads stay before WMMAs

        // 16x128 @ 128x16 tile, two independent accumulator chains
        v8f c0 = {}, c1 = {};
        c0 = __builtin_amdgcn_wmma_f32_16x16x32_f16(false, afr[0], false, bf[0], (short)0, c0, false, false);
        c1 = __builtin_amdgcn_wmma_f32_16x16x32_f16(false, afr[1], false, bf[1], (short)0, c1, false, false);
        c0 = __builtin_amdgcn_wmma_f32_16x16x32_f16(false, afr[2], false, bf[2], (short)0, c0, false, false);
        c1 = __builtin_amdgcn_wmma_f32_16x16x32_f16(false, afr[3], false, bf[3], (short)0, c1, false, false);

        // alpha update with sequence mask
#pragma unroll
        for (int r = 0; r < 8; ++r) {
            int m = r + 8 * lh;
            float s   = fmaxf(c0[r] + c1[r], 1e-30f);
            float nxt = fmaf(__builtin_amdgcn_logf(s), LN2, mrow_s[m] + ev[r]);
            float old = alpha_s[m * T + jj];
            alpha_s[m * T + jj] = (t < len_s[m]) ? nxt : old;
        }
        __syncthreads();
    }

    // logZ[b] = logsumexp_j(alpha[b][j] + end[j])
    float mx = -INFINITY;
#pragma unroll
    for (int k2 = 0; k2 < 8; ++k2) {
        int j = jl + 16 * k2;
        mx = fmaxf(mx, alpha_s[m_a * T + j] + end_t[j]);
    }
#pragma unroll
    for (int off = 8; off; off >>= 1)
        mx = fmaxf(mx, __shfl_xor(mx, off, 32));
    float sm = 0.f;
#pragma unroll
    for (int k2 = 0; k2 < 8; ++k2) {
        int j = jl + 16 * k2;
        sm += fast_exp(alpha_s[m_a * T + j] + end_t[j] - mx);
    }
#pragma unroll
    for (int off = 8; off; off >>= 1)
        sm += __shfl_xor(sm, off, 32);
    if (jl == 0) logZ[b0 + m_a] = mx + fast_log(sm);
}

// ---------------------------------------------------------------------------
// Viterbi scan (max-plus): one block per batch, transitions in LDS (64 KB).
// 512 threads: thread (part, j) scans i in [32*part, 32*part+32).
// ---------------------------------------------------------------------------
__global__ __launch_bounds__(512) void crf_viterbi(
    const float* __restrict__ logit, const int* __restrict__ seq_lens,
    const float* __restrict__ trans, const float* __restrict__ start_t,
    const float* __restrict__ end_t,
    unsigned char* __restrict__ hist, int* __restrict__ last_tag)
{
    __shared__ float trans_s[T * T];   // 64 KB (CDNA5: up to 320 KB LDS/WG)
    __shared__ float score_s[T];
    __shared__ float candv[512];
    __shared__ int   candi[512];

    const int b = blockIdx.x, tid = threadIdx.x;
    for (int q = tid; q < T * T; q += 512) trans_s[q] = trans[q];
    const int len = seq_lens[b];
    if (tid < T) score_s[tid] = start_t[tid] + logit[(size_t)b * S * T + tid];
    __syncthreads();

    const int j = tid & 127, part = tid >> 7;   // part in 0..3
    for (int t = 1; t < S; ++t) {
        float best = -INFINITY; int bi = 0;
        const int i0 = part * 32;
#pragma unroll 8
        for (int i = i0; i < i0 + 32; ++i) {
            float v = score_s[i] + trans_s[i * T + j];
            if (v > best) { best = v; bi = i; }     // strict > keeps first argmax
        }
        candv[tid] = best; candi[tid] = bi;
        __syncthreads();
        if (tid < T) {
            float bv = candv[tid]; int ii = candi[tid];
#pragma unroll
            for (int p = 1; p < 4; ++p) {
                float b2 = candv[tid + 128 * p]; int i2 = candi[tid + 128 * p];
                if (b2 > bv) { bv = b2; ii = i2; }  // tie -> lower i (lower part)
            }
            bool act = (t < len);
            float e = logit[((size_t)b * S + t) * T + j];
            int   idx = act ? ii : j;
            float ns  = act ? (bv + e) : score_s[j];
            hist[((size_t)b * (S - 1) + (t - 1)) * T + j] = (unsigned char)idx;
            score_s[j] = ns;
        }
        __syncthreads();
    }
    // last_tag = argmax_j(score + end)
    if (tid < T) { candv[tid] = score_s[tid] + end_t[tid]; candi[tid] = tid; }
    __syncthreads();
    for (int off = 64; off >= 1; off >>= 1) {
        if (tid < off && candv[tid + off] > candv[tid]) {
            candv[tid] = candv[tid + off]; candi[tid] = candi[tid + off];
        }
        __syncthreads();
    }
    if (tid == 0) last_tag[b] = candi[0];
}

// ---------------------------------------------------------------------------
// Backtrace: one lane per batch, chain through batch-major hist (L2-resident).
// ---------------------------------------------------------------------------
__global__ void crf_backtrace(const unsigned char* __restrict__ hist,
                              const int* __restrict__ last_tag,
                              float* __restrict__ predf, int* __restrict__ pred_int)
{
    int b = blockIdx.x * blockDim.x + threadIdx.x;
    if (b >= B) return;
    int tag = last_tag[b];
    predf[(size_t)b * S + (S - 1)] = (float)tag;
    pred_int[b * S + (S - 1)] = tag;
    for (int t = S - 2; t >= 0; --t) {
        tag = hist[((size_t)b * (S - 1) + t) * T + tag];
        predf[(size_t)b * S + t] = (float)tag;
        pred_int[b * S + t] = tag;
    }
}

// ---------------------------------------------------------------------------
// Numerator: one block per batch.
// ---------------------------------------------------------------------------
__global__ __launch_bounds__(128) void crf_num(
    const float* __restrict__ logit, const int* __restrict__ target,
    const int* __restrict__ seq_lens, const float* __restrict__ trans,
    const float* __restrict__ start_t, const float* __restrict__ end_t,
    float* __restrict__ num)
{
    __shared__ float red[128];
    const int b = blockIdx.x, tid = threadIdx.x;
    const int len = seq_lens[b];
    float acc = 0.f;
    for (int t = tid; t < S; t += 128) {
        int tg = target[b * S + t];
        float em = logit[((size_t)b * S + t) * T + tg];
        if (t == 0) acc += start_t[tg] + em;
        else if (t < len) acc += em + trans[target[b * S + t - 1] * T + tg];
    }
    red[tid] = acc; __syncthreads();
    for (int off = 64; off >= 1; off >>= 1) {
        if (tid < off) red[tid] += red[tid + off];
        __syncthreads();
    }
    if (tid == 0) num[b] = red[0] + end_t[target[b * S + len - 1]];
}

// ---------------------------------------------------------------------------
// Swapped log-softmax: swap permutes two entries => max & sumexp unchanged,
// so log_softmax(swapped) = swapped - logsumexp(original row).
// One wave per (b,s) row, float4 per lane.
// ---------------------------------------------------------------------------
__global__ __launch_bounds__(256) void crf_logprobs(
    const float* __restrict__ logit, const int* __restrict__ pred_int,
    float* __restrict__ logp)
{
    const int lane = threadIdx.x & 31, wv = threadIdx.x >> 5;
    const size_t row = (size_t)blockIdx.x * 8 + wv;
    const float* rp = logit + row * T;
    float4 x = ((const float4*)rp)[lane];
    float mv = x.x; int mi = lane * 4;
    if (x.y > mv) { mv = x.y; mi = lane * 4 + 1; }
    if (x.z > mv) { mv = x.z; mi = lane * 4 + 2; }
    if (x.w > mv) { mv = x.w; mi = lane * 4 + 3; }
#pragma unroll
    for (int off = 16; off >= 1; off >>= 1) {
        float ov = __shfl_xor(mv, off, 32);
        int   oi = __shfl_xor(mi, off, 32);
        if (ov > mv || (ov == mv && oi < mi)) { mv = ov; mi = oi; }
    }
    float se = fast_exp(x.x - mv) + fast_exp(x.y - mv) +
               fast_exp(x.z - mv) + fast_exp(x.w - mv);
#pragma unroll
    for (int off = 16; off >= 1; off >>= 1) se += __shfl_xor(se, off, 32);
    const int pr = pred_int[row];
    const float vv = rp[pr];
    const float lse = mv + fast_log(se);
    float4 o;
    {
        int j0 = lane * 4;
        float a0 = (j0     == pr) ? mv : ((j0     == mi) ? vv : x.x);
        float a1 = (j0 + 1 == pr) ? mv : ((j0 + 1 == mi) ? vv : x.y);
        float a2 = (j0 + 2 == pr) ? mv : ((j0 + 2 == mi) ? vv : x.z);
        float a3 = (j0 + 3 == pr) ? mv : ((j0 + 3 == mi) ? vv : x.w);
        o = make_float4(a0 - lse, a1 - lse, a2 - lse, a3 - lse);
    }
    ((float4*)(logp + row * T))[lane] = o;
}

__global__ void crf_loss(const float* __restrict__ num,
                         const float* __restrict__ logZ, float* __restrict__ out0)
{
    __shared__ float red[128];
    int tid = threadIdx.x;
    red[tid] = num[tid] - logZ[tid];
    __syncthreads();
    for (int off = 64; off >= 1; off >>= 1) {
        if (tid < off) red[tid] += red[tid + off];
        __syncthreads();
    }
    if (tid == 0) out0[0] = -red[0] / (float)B;
}

// ---------------------------------------------------------------------------
extern "C" void kernel_launch(void* const* d_in, const int* in_sizes, int n_in,
                              void* d_out, int out_size, void* d_ws, size_t ws_size,
                              hipStream_t stream) {
    const float* logit   = (const float*)d_in[0];
    const int*   target  = (const int*)d_in[1];
    const int*   seq_lens= (const int*)d_in[2];
    const float* trans   = (const float*)d_in[3];
    const float* start_t = (const float*)d_in[4];
    const float* end_t   = (const float*)d_in[5];

    float* out   = (float*)d_out;
    float* loss  = out;
    float* predf = out + 1;
    float* logp  = out + 1 + (size_t)B * S;

    char* ws = (char*)d_ws;
    const size_t HB = (size_t)B * (S - 1) * T;       // 33.5 MB uint8 history
    unsigned char* hist = (unsigned char*)ws;
    float* logZ     = (float*)(ws + HB);
    float* num      = logZ + B;
    int*   last_tag = (int*)(num + B);
    int*   pred_int = last_tag + B;

    crf_forward  <<<B / 16, 256, 0, stream>>>(logit, seq_lens, trans, start_t, end_t, logZ);
    crf_viterbi  <<<B,      512, 0, stream>>>(logit, seq_lens, trans, start_t, end_t, hist, last_tag);
    crf_num      <<<B,      128, 0, stream>>>(logit, target, seq_lens, trans, start_t, end_t, num);
    crf_backtrace<<<4,       32, 0, stream>>>(hist, last_tag, predf, pred_int);
    crf_logprobs <<<(B * S) / 8, 256, 0, stream>>>(logit, pred_int, logp);
    crf_loss     <<<1,      128, 0, stream>>>(num, logZ, loss);
}
